// QuillanSOTA_798863917299
// MI455X (gfx1250) — matrix operand order repaired and checked by reading, not compiled, not run on hardware
//
#include <hip/hip_runtime.h>
#include <hip/hip_bf16.h>

#define DIM      512
#define NLAYERS  6
#define VOCABSZ  50257
#define HEADS    8
#define EXPERTS  8
#define SEQ      1024
#define BATCH    2
#define NTOK     (BATCH * SEQ)
#define QKVW     (3 * DIM)      // 1536
#define FFDIM    (2 * DIM)      // 1024

typedef __bf16 bf16_t;
typedef __attribute__((ext_vector_type(16))) __bf16 bf16x16;
typedef __attribute__((ext_vector_type(8)))  float  f32x8;

union FragAB {
    bf16x16 v;
    uint4   q[2];
    __bf16  h[16];
};

__device__ __forceinline__ f32x8 wmma_bf16(bf16x16 a, bf16x16 b, f32x8 c) {
    return __builtin_amdgcn_wmma_f32_16x16x32_bf16(false, a, false, b, (short)0, c, false, false);
}

__device__ __forceinline__ f32x8 zero8() {
    f32x8 z = {0.f, 0.f, 0.f, 0.f, 0.f, 0.f, 0.f, 0.f};
    return z;
}

// Issue an async global->LDS 16-byte copy (gfx1250 TDM-adjacent path, ASYNCcnt).
__device__ __forceinline__ void async_copy_b128(const bf16_t* gptr, __bf16* lptr) {
    const unsigned lds = (unsigned)(size_t)lptr;   // low 32 bits of LDS aperture addr = LDS offset
    asm volatile("global_load_async_to_lds_b128 %0, %1, off"
                 :: "v"(lds), "v"(gptr) : "memory");
}

// ---------------------------------------------------------------------------
// Generic bf16 WMMA GEMM: C[N,M] = epilogue(A[N,K] @ W[M,K]^T)
// flags: bit0 = exact GELU, bit1 = accumulate into Cf
// Block = 4 waves, 64x64 output tile (each wave 16 rows x 64 cols, 4 accums).
// W tile (64 cols x 32 K) staged in LDS via async copies, double buffered,
// shared across the 4 waves. Requires Nrows % 64 == 0 (true for all calls).
// ---------------------------------------------------------------------------
__global__ __launch_bounds__(128)
void wmma_gemm_bf16(const bf16_t* __restrict__ A, const bf16_t* __restrict__ W,
                    int Nrows, int M, int K,
                    const float* __restrict__ bias,
                    const float* __restrict__ rowscale, int rs_stride,
                    const float* __restrict__ residual,
                    float* __restrict__ Cf, bf16_t* __restrict__ Cbf,
                    int flags)
{
    __shared__ __attribute__((aligned(16))) __bf16 Wtile[2][64 * 32];   // 2 x 4 KB

    const int tid  = threadIdx.x;
    const int lane = tid & 31;
    const int wave = tid >> 5;
    const int row0 = blockIdx.y * 64 + wave * 16;
    const int col0 = blockIdx.x * 64;

    const int half = lane >> 4;          // 0: lanes 0-15, 1: lanes 16-31
    const int r    = lane & 15;
    const int selA = half * 8;           // A frag K-interleave (ISA 16x32 bf16 A layout)
    const int selB = half * 8;           // B frag K-interleave (matches A)

    // Staging geometry: 256 x 16B chunks per tile, 2 per thread.
    const int chunk0 = tid;              // chunk -> col = chunk>>2, part = chunk&3
    const int nk = K / 32;

    f32x8 acc[4];
    #pragma unroll
    for (int j = 0; j < 4; ++j) acc[j] = zero8();

    const bf16_t* arow = A + (size_t)(row0 + r) * K;

    // Prologue: stage K-step 0 into buffer 0.
    #pragma unroll
    for (int c = 0; c < 2; ++c) {
        const int chunk = chunk0 + c * 128;
        const int col   = chunk >> 2;
        const int part  = chunk & 3;
        const int wr    = min(col0 + col, M - 1);     // clamp: garbage cols never stored
        async_copy_b128(W + (size_t)wr * K + part * 8,
                        &Wtile[0][col * 32 + part * 8]);
    }

    for (int ks = 0; ks < nk; ++ks) {
        const int cur = ks & 1;
        if (ks + 1 < nk) {
            const int knext = (ks + 1) * 32;
            #pragma unroll
            for (int c = 0; c < 2; ++c) {
                const int chunk = chunk0 + c * 128;
                const int col   = chunk >> 2;
                const int part  = chunk & 3;
                const int wr    = min(col0 + col, M - 1);
                async_copy_b128(W + (size_t)wr * K + knext + part * 8,
                                &Wtile[cur ^ 1][col * 32 + part * 8]);
            }
            asm volatile("s_wait_asynccnt 0x2" ::: "memory");  // current buffer landed
        } else {
            asm volatile("s_wait_asynccnt 0x0" ::: "memory");
        }
        __syncthreads();     // all waves' portions visible

        const int k = ks * 32;
        FragAB a;
        a.q[0] = *(const uint4*)(arow + k + selA);
        a.q[1] = *(const uint4*)(arow + k + 16 + selA);
        #pragma unroll
        for (int j = 0; j < 4; ++j) {
            FragAB b;
            const __bf16* wp = &Wtile[cur][(j * 16 + r) * 32 + selB];
            b.q[0] = *(const uint4*)(wp);
            b.q[1] = *(const uint4*)(wp + 16);
            acc[j] = wmma_bf16(a.v, b.v, acc[j]);
        }
        __syncthreads();     // done reading before this buffer is rewritten
    }

    // Epilogue. C frag: vgpr i -> row m = i + 8*half, col n = r + 16*j.
    #pragma unroll
    for (int j = 0; j < 4; ++j) {
        const int col = col0 + j * 16 + r;
        if (col >= M) continue;
        #pragma unroll
        for (int i = 0; i < 8; ++i) {
            const int row = row0 + i + half * 8;
            float c = acc[j][i];
            if (bias)      c += bias[col];
            if (flags & 1) c = 0.5f * c * (1.0f + erff(c * 0.70710678118654752f));
            if (rowscale)  c *= rowscale[(size_t)row * rs_stride];
            const size_t o = (size_t)row * M + col;
            if (residual)  c += residual[o];
            if (flags & 2) c += Cf[o];
            if (Cf)  Cf[o]  = c;
            if (Cbf) Cbf[o] = (bf16_t)c;
        }
    }
}

// ---------------------------------------------------------------------------
// Flash attention, WMMA, one wave per 16 query rows of one (b, h).
// qkv: bf16 [NTOK][1536], layout col = c*512 + h*64 + d.
// out: bf16 [NTOK][512]  (attention output, pre out-projection)
// ---------------------------------------------------------------------------
__global__ __launch_bounds__(128)
void attention_kernel(const bf16_t* __restrict__ qkv, bf16_t* __restrict__ out)
{
    __shared__ __attribute__((aligned(16))) __bf16 Ptile[4][16 * 32];

    const int lane = threadIdx.x & 31;
    const int wave = threadIdx.x >> 5;
    const int gid  = blockIdx.x * 4 + wave;      // 1024 waves total
    const int qb   = gid & 63;
    const int h    = (gid >> 6) & 7;
    const int b    = gid >> 9;

    const int half = lane >> 4;
    const int r    = lane & 15;
    const int selA = half * 8;
    const int selB = half * 8;

    const size_t base = (size_t)b * SEQ * QKVW;
    const bf16_t* Q  = qkv + base + h * 64;
    const bf16_t* Kp = qkv + base + DIM + h * 64;
    const bf16_t* Vp = qkv + base + 2 * DIM + h * 64;

    // Q A-fragments (hd=64 -> two K=32 steps), rows qb*16 + r
    const bf16_t* qrow = Q + (size_t)(qb * 16 + r) * QKVW;
    FragAB qf0, qf1;
    qf0.q[0] = *(const uint4*)(qrow + selA);
    qf0.q[1] = *(const uint4*)(qrow + 16 + selA);
    qf1.q[0] = *(const uint4*)(qrow + 32 + selA);
    qf1.q[1] = *(const uint4*)(qrow + 48 + selA);

    f32x8 o[4];
    #pragma unroll
    for (int j = 0; j < 4; ++j) o[j] = zero8();
    float mrun[8], lrun[8];
    #pragma unroll
    for (int i = 0; i < 8; ++i) { mrun[i] = -1e30f; lrun[i] = 0.f; }

    for (int kb = 0; kb < SEQ; kb += 32) {
        // S = Q @ K^T * scale : two 16x16 tiles (key cols kb..+15, kb+16..+31)
        f32x8 s0 = zero8(), s1 = zero8();
        #pragma unroll
        for (int s = 0; s < 2; ++s) {
            FragAB kf0, kf1;
            const bf16_t* k0 = Kp + (size_t)(kb + r) * QKVW + s * 32 + selB;
            const bf16_t* k1 = Kp + (size_t)(kb + 16 + r) * QKVW + s * 32 + selB;
            kf0.q[0] = *(const uint4*)(k0); kf0.q[1] = *(const uint4*)(k0 + 16);
            kf1.q[0] = *(const uint4*)(k1); kf1.q[1] = *(const uint4*)(k1 + 16);
            s0 = wmma_bf16(s ? qf1.v : qf0.v, kf0.v, s0);
            s1 = wmma_bf16(s ? qf1.v : qf0.v, kf1.v, s1);
        }

        // Online softmax; row m = i + 8*half, its 16 cols live across a
        // 16-lane half-wave -> width-16 shfl_xor butterfly reductions.
        #pragma unroll
        for (int i = 0; i < 8; ++i) {
            float a0 = s0[i] * 0.125f;      // hd^-0.5
            float a1 = s1[i] * 0.125f;
            float mx = fmaxf(a0, a1);
            mx = fmaxf(mx, __shfl_xor(mx, 1, 32));
            mx = fmaxf(mx, __shfl_xor(mx, 2, 32));
            mx = fmaxf(mx, __shfl_xor(mx, 4, 32));
            mx = fmaxf(mx, __shfl_xor(mx, 8, 32));
            const float mnew  = fmaxf(mrun[i], mx);
            const float alpha = __expf(mrun[i] - mnew);
            const float p0 = __expf(a0 - mnew);
            const float p1 = __expf(a1 - mnew);
            float rs = p0 + p1;
            rs += __shfl_xor(rs, 1, 32);
            rs += __shfl_xor(rs, 2, 32);
            rs += __shfl_xor(rs, 4, 32);
            rs += __shfl_xor(rs, 8, 32);
            lrun[i] = lrun[i] * alpha + rs;
            mrun[i] = mnew;
            #pragma unroll
            for (int j = 0; j < 4; ++j) o[j][i] *= alpha;
            const int prow = i + half * 8;
            Ptile[wave][prow * 32 + r]      = (bf16_t)p0;
            Ptile[wave][prow * 32 + 16 + r] = (bf16_t)p1;
        }

        // Reload P (16x32) as A-fragment from LDS (same-wave DS in-order).
        FragAB pf;
        pf.q[0] = *(const uint4*)(&Ptile[wave][r * 32 + selA]);
        pf.q[1] = *(const uint4*)(&Ptile[wave][r * 32 + 16 + selA]);

        // O += P @ V  (K = 32 keys; B frag: lane col = hd idx, keys interleaved like A)
        #pragma unroll
        for (int j = 0; j < 4; ++j) {
            FragAB vf;
            #pragma unroll
            for (int t = 0; t < 16; ++t)
                vf.h[t] = Vp[(size_t)(kb + selB + t + (t >= 8 ? 8 : 0)) * QKVW + j * 16 + r];
            o[j] = wmma_bf16(pf.v, vf.v, o[j]);
        }
    }

    // Normalize and store: out[(b*SEQ + row)*512 + h*64 + col]
    #pragma unroll
    for (int i = 0; i < 8; ++i) {
        const int row = qb * 16 + i + half * 8;
        const float invl = 1.0f / lrun[i];
        #pragma unroll
        for (int j = 0; j < 4; ++j)
            out[((size_t)(b * SEQ + row)) * DIM + h * 64 + j * 16 + r] =
                (bf16_t)(o[j][i] * invl);
    }
}

// ---------------------------------------------------------------------------
// Helper kernels
// ---------------------------------------------------------------------------
__global__ void embed_kernel(const int* __restrict__ ids, const float* __restrict__ emb,
                             float* __restrict__ X32, bf16_t* __restrict__ XBF)
{
    const int t  = blockIdx.x;
    const int id = ids[t];
    #pragma unroll
    for (int i = 0; i < 2; ++i) {
        const int d = threadIdx.x + 256 * i;
        const float v = emb[(size_t)id * DIM + d];
        X32[(size_t)t * DIM + d] = v;
        XBF[(size_t)t * DIM + d] = (bf16_t)v;
    }
}

__global__ void cvt_kernel(const float* __restrict__ src, bf16_t* __restrict__ dst, size_t n)
{
    const size_t i = (size_t)blockIdx.x * 256 + threadIdx.x;
    if (i < n) dst[i] = (bf16_t)src[i];
}

// BitNet 1.58b: scale = max(mean|w| per row, 1e-5); w = clip(round(w/scale),-1,1)*scale
__global__ __launch_bounds__(256)
void bitnet_kernel(const float* __restrict__ w, bf16_t* __restrict__ out, int K)
{
    __shared__ float red[8];
    const int row = blockIdx.x, tid = threadIdx.x;
    const int lane = tid & 31, wv = tid >> 5;
    const float* wr = w + (size_t)row * K;

    float s = 0.f;
    for (int k = tid; k < K; k += 256) s += fabsf(wr[k]);
    s += __shfl_xor(s, 16, 32); s += __shfl_xor(s, 8, 32);
    s += __shfl_xor(s, 4, 32);  s += __shfl_xor(s, 2, 32);
    s += __shfl_xor(s, 1, 32);
    if (lane == 0) red[wv] = s;
    __syncthreads();
    if (tid == 0) {
        float t2 = 0.f;
        #pragma unroll
        for (int i = 0; i < 8; ++i) t2 += red[i];
        red[0] = t2;
    }
    __syncthreads();
    const float scale = fmaxf(red[0] / (float)K, 1e-5f);
    const float inv = 1.0f / scale;
    for (int k = tid; k < K; k += 256) {
        float q = rintf(wr[k] * inv);
        q = fminf(1.f, fmaxf(-1.f, q));
        out[(size_t)row * K + k] = (bf16_t)(q * scale);
    }
}

// Router: softmax + top-2 + renorm -> combine weights cw[N][8]
__global__ __launch_bounds__(32)
void router_kernel(const float* __restrict__ X, const float* __restrict__ rw,
                   const float* __restrict__ rb, float* __restrict__ cw)
{
    const int t = blockIdx.x, lane = threadIdx.x;
    const float* x = X + (size_t)t * DIM;
    float acc[EXPERTS];
    #pragma unroll
    for (int e = 0; e < EXPERTS; ++e) acc[e] = 0.f;
    for (int d = lane; d < DIM; d += 32) {
        const float xv = x[d];
        #pragma unroll
        for (int e = 0; e < EXPERTS; ++e) acc[e] += xv * rw[e * DIM + d];
    }
    #pragma unroll
    for (int e = 0; e < EXPERTS; ++e) {
        float v = acc[e];
        v += __shfl_xor(v, 16, 32); v += __shfl_xor(v, 8, 32);
        v += __shfl_xor(v, 4, 32);  v += __shfl_xor(v, 2, 32);
        v += __shfl_xor(v, 1, 32);
        acc[e] = v + rb[e];
    }
    float mx = acc[0];
    #pragma unroll
    for (int e = 1; e < EXPERTS; ++e) mx = fmaxf(mx, acc[e]);
    float p[EXPERTS], den = 0.f;
    #pragma unroll
    for (int e = 0; e < EXPERTS; ++e) { p[e] = __expf(acc[e] - mx); den += p[e]; }
    #pragma unroll
    for (int e = 0; e < EXPERTS; ++e) p[e] /= den;
    int i1 = 0; float v1 = p[0];
    #pragma unroll
    for (int e = 1; e < EXPERTS; ++e) if (p[e] > v1) { v1 = p[e]; i1 = e; }
    int i2 = -1; float v2 = -1.f;
    #pragma unroll
    for (int e = 0; e < EXPERTS; ++e) if (e != i1 && p[e] > v2) { v2 = p[e]; i2 = e; }
    const float snorm = v1 + v2 + 1e-8f;
    if (lane < EXPERTS)
        cw[(size_t)t * EXPERTS + lane] =
            (lane == i1) ? v1 / snorm : (lane == i2) ? v2 / snorm : 0.f;
}

// LayerNorm(moe + residual) -> X32 (fp32 master) + XBF (bf16)
__global__ __launch_bounds__(32)
void ln_kernel(const float* __restrict__ moe, const float* __restrict__ res,
               const float* __restrict__ g, const float* __restrict__ bb,
               float* __restrict__ X32, bf16_t* __restrict__ XBF)
{
    const int t = blockIdx.x, lane = threadIdx.x;
    const size_t base = (size_t)t * DIM;
    float v[16], sum = 0.f;
    #pragma unroll
    for (int i = 0; i < 16; ++i) {
        const int d = lane + 32 * i;
        v[i] = moe[base + d] + res[base + d];
        sum += v[i];
    }
    sum += __shfl_xor(sum, 16, 32); sum += __shfl_xor(sum, 8, 32);
    sum += __shfl_xor(sum, 4, 32);  sum += __shfl_xor(sum, 2, 32);
    sum += __shfl_xor(sum, 1, 32);
    const float mu = sum / (float)DIM;
    float var = 0.f;
    #pragma unroll
    for (int i = 0; i < 16; ++i) { const float d2 = v[i] - mu; var += d2 * d2; }
    var += __shfl_xor(var, 16, 32); var += __shfl_xor(var, 8, 32);
    var += __shfl_xor(var, 4, 32);  var += __shfl_xor(var, 2, 32);
    var += __shfl_xor(var, 1, 32);
    const float rstd = rsqrtf(var / (float)DIM + 1e-5f);
    #pragma unroll
    for (int i = 0; i < 16; ++i) {
        const int d = lane + 32 * i;
        const float y = (v[i] - mu) * rstd * g[d] + bb[d];
        X32[base + d] = y;
        XBF[base + d] = (bf16_t)y;
    }
}

// ---------------------------------------------------------------------------
// Host side
// ---------------------------------------------------------------------------
static void launch_gemm(hipStream_t st, const bf16_t* A, const bf16_t* W,
                        int N, int M, int K,
                        const float* bias, const float* rowscale, int rs_stride,
                        const float* residual, float* Cf, bf16_t* Cbf, int flags)
{
    dim3 grid((M + 63) / 64, (N + 63) / 64);
    wmma_gemm_bf16<<<grid, dim3(128), 0, st>>>(A, W, N, M, K, bias, rowscale,
                                               rs_stride, residual, Cf, Cbf, flags);
}

extern "C" void kernel_launch(void* const* d_in, const int* in_sizes, int n_in,
                              void* d_out, int out_size, void* d_ws, size_t ws_size,
                              hipStream_t stream)
{
    (void)in_sizes; (void)n_in; (void)out_size; (void)ws_size;
    const int*   ids      = (const int*)  d_in[0];
    const float* embed    = (const float*)d_in[1];
    const float* qkv_w    = (const float*)d_in[2];
    const float* out_w    = (const float*)d_in[3];
    const float* router_w = (const float*)d_in[4];
    const float* router_b = (const float*)d_in[5];
    const float* w1       = (const float*)d_in[6];
    const float* b1       = (const float*)d_in[7];
    const float* w2       = (const float*)d_in[8];
    const float* b2       = (const float*)d_in[9];
    const float* ln_g     = (const float*)d_in[10];
    const float* ln_b     = (const float*)d_in[11];
    const float* head_w   = (const float*)d_in[12];
    float* out = (float*)d_out;

    char* ws = (char*)d_ws;
    size_t off = 0;
    auto alloc = [&](size_t bytes) -> char* {
        char* p = ws + off;
        off = (off + bytes + 255) & ~(size_t)255;
        return p;
    };
    float*  X32  = (float*) alloc((size_t)NTOK * DIM * 4);
    bf16_t* XBF  = (bf16_t*)alloc((size_t)NTOK * DIM * 2);
    bf16_t* QKV  = (bf16_t*)alloc((size_t)NTOK * QKVW * 2);
    bf16_t* ATTN = (bf16_t*)alloc((size_t)NTOK * DIM * 2);
    float*  MOE  = (float*) alloc((size_t)NTOK * DIM * 4);
    bf16_t* HBUF = (bf16_t*)alloc((size_t)NTOK * FFDIM * 2);
    float*  CW   = (float*) alloc((size_t)NTOK * EXPERTS * 4);
    bf16_t* W2B  = (bf16_t*)alloc((size_t)EXPERTS * DIM * FFDIM * 2);
    bf16_t* WB   = (bf16_t*)alloc((size_t)VOCABSZ * DIM * 2);  // reused staging

    embed_kernel<<<NTOK, 256, 0, stream>>>(ids, embed, X32, XBF);

    for (int l = 0; l < NLAYERS; ++l) {
        // QKV projection
        const size_t nq = (size_t)QKVW * DIM;
        cvt_kernel<<<(nq + 255) / 256, 256, 0, stream>>>(qkv_w + (size_t)l * nq, WB, nq);
        launch_gemm(stream, XBF, WB, NTOK, QKVW, DIM,
                    nullptr, nullptr, 0, nullptr, nullptr, QKV, 0);

        // Attention (1024 waves, 4 per block)
        attention_kernel<<<256, 128, 0, stream>>>(QKV, ATTN);

        // Out projection + residual: x = x + attn @ out_w^T
        const size_t no = (size_t)DIM * DIM;
        cvt_kernel<<<(no + 255) / 256, 256, 0, stream>>>(out_w + (size_t)l * no, WB, no);
        launch_gemm(stream, ATTN, WB, NTOK, DIM, DIM,
                    nullptr, nullptr, 0, X32, X32, XBF, 0);

        // Router: combine weights
        router_kernel<<<NTOK, 32, 0, stream>>>(X32,
            router_w + (size_t)l * EXPERTS * DIM, router_b + (size_t)l * EXPERTS, CW);

        // BitNet-quantize expert weights to bf16
        bitnet_kernel<<<EXPERTS * FFDIM, 256, 0, stream>>>(
            w1 + (size_t)l * EXPERTS * FFDIM * DIM, WB, DIM);
        bitnet_kernel<<<EXPERTS * DIM, 256, 0, stream>>>(
            w2 + (size_t)l * EXPERTS * DIM * FFDIM, W2B, FFDIM);

        // Dense MoE, expert-sequential accumulation
        for (int e = 0; e < EXPERTS; ++e) {
            launch_gemm(stream, XBF, WB + (size_t)e * FFDIM * DIM, NTOK, FFDIM, DIM,
                        b1 + ((size_t)l * EXPERTS + e) * FFDIM,
                        nullptr, 0, nullptr, nullptr, HBUF, /*gelu*/1);
            launch_gemm(stream, HBUF, W2B + (size_t)e * DIM * FFDIM, NTOK, DIM, FFDIM,
                        b2 + ((size_t)l * EXPERTS + e) * DIM,
                        CW + e, EXPERTS, nullptr, MOE, nullptr, e ? /*acc*/2 : 0);
        }

        // LayerNorm(moe + x)
        ln_kernel<<<NTOK, 32, 0, stream>>>(MOE, X32,
            ln_g + (size_t)l * DIM, ln_b + (size_t)l * DIM, X32, XBF);
    }

    // Head: logits = x @ head_w^T  (fp32 out, ragged M handled in-kernel)
    const size_t nh = (size_t)VOCABSZ * DIM;
    cvt_kernel<<<(nh + 255) / 256, 256, 0, stream>>>(head_w, WB, nh);
    launch_gemm(stream, XBF, WB, NTOK, VOCABSZ, DIM,
                nullptr, nullptr, 0, nullptr, out, nullptr, 0);
}
